// MultiStateAttention_66314295050939
// MI455X (gfx1250) — 16-bit and 8-bit matrix operand order repaired and checked by reading, compiled, not run on hardware
//
#include <hip/hip_runtime.h>
#include <hip/hip_bf16.h>

typedef __attribute__((ext_vector_type(16))) _Float16 v16h;
typedef __attribute__((ext_vector_type(8)))  _Float16 v8h;
typedef __attribute__((ext_vector_type(8)))  float    v8f;
typedef __attribute__((ext_vector_type(8)))  int      v8i;

namespace {
constexpr int Bn = 8;
constexpr int Ln = 1024;
constexpr int Cn = 512;
constexpr int Hn = 8;
constexpr int Dn = 64;
}

__device__ __forceinline__ v16h cat16(v8h lo, v8h hi) {
  v16h r;
#pragma unroll
  for (int i = 0; i < 8; ++i) { r[i] = lo[i]; r[i + 8] = hi[i]; }
  return r;
}


__global__ __launch_bounds__(256) void cvt_f32_to_f16_kernel(
    const float* __restrict__ in, _Float16* __restrict__ out, int n) {
  int t = blockIdx.x * blockDim.x + threadIdx.x;
  if (t < n) out[t] = (_Float16)in[t];
}

__global__ __launch_bounds__(256) void transpose_to_f16_kernel(
    const float* __restrict__ w, _Float16* __restrict__ wT, int K, int N) {
  int t = blockIdx.x * blockDim.x + threadIdx.x;
  if (t >= K * N) return;
  int n = t % N, k = t / N;
  wT[(size_t)n * K + k] = (_Float16)w[t];
}

__global__ __launch_bounds__(256) void pack_adj_kernel(
    const float* __restrict__ adj, unsigned char* __restrict__ adj8, int n) {
  int t = blockIdx.x * blockDim.x + threadIdx.x;
  if (t >= n) return;
  int j = t & (Ln - 1);
  int i = (t >> 10) & (Ln - 1);
  adj8[t] = (unsigned char)((i != 0 && j != 0 && adj[t] != 0.0f) ? 1 : 0);
}

__global__ __launch_bounds__(128) void struct_bias_kernel(
    const unsigned char* __restrict__ adj8,
    const float* __restrict__ distance,
    unsigned char* __restrict__ bias_pack) {
  int wave = (blockIdx.x * blockDim.x + threadIdx.x) >> 5;
  int lane = threadIdx.x & 31;
  int jg = wave & 31;
  int it = (wave >> 5) & 63;
  int b = wave >> 11;
  if (b >= Bn) return;

  const unsigned char* ab = adj8 + (size_t)b * Ln * Ln;
  const int lan = lane & 15;
  const bool hiL = lane >= 16;
  const int rowA = it * 16 + lan;
  const int rowB0 = jg * 32 + lan;
  const int rowB1 = rowB0 + 16;

  v8i acc0 = {}, acc1 = {};
  for (int k0 = 0; k0 < Ln; k0 += 64) {
    v8i a, bm0, bm1;
#pragma unroll
    for (int d = 0; d < 8; ++d) {
      int ka = k0 + (d >> 2) * 32 + (hiL ? 16 : 0) + (d & 3) * 4;
      a[d] = *(const int*)(ab + (size_t)rowA * Ln + ka);
      int kb = k0 + (d >> 2) * 32 + (hiL ? 16 : 0) + (d & 3) * 4;
      bm0[d] = *(const int*)(ab + (size_t)rowB0 * Ln + kb);
      bm1[d] = *(const int*)(ab + (size_t)rowB1 * Ln + kb);
    }
    acc0 = __builtin_amdgcn_wmma_i32_16x16x64_iu8(false, a, false, bm0, acc0, false, false);
    acc1 = __builtin_amdgcn_wmma_i32_16x16x64_iu8(false, a, false, bm1, acc1, false, false);
  }

  const int hi = hiL ? 8 : 0;
  const float* db = distance + (size_t)b * Ln * Ln;
  unsigned char* bp = bias_pack + (size_t)b * Ln * Ln;
#pragma unroll
  for (int t = 0; t < 2; ++t) {
    int j = jg * 32 + t * 16 + lan;
#pragma unroll
    for (int r = 0; r < 8; ++r) {
      int i = it * 16 + r + hi;
      int sv = (t == 0) ? acc0[r] : acc1[r];
      unsigned m = 0;
      if (ab[(size_t)i * Ln + j]) m |= 1u;
      if (ab[(size_t)j * Ln + i]) m |= 2u;
      if (db[(size_t)i * Ln + j] > 0.0f) m |= 4u;
      if (db[(size_t)j * Ln + i] > 0.0f) m |= 8u;
      if (sv > 0) m |= 16u;
      bp[(size_t)i * Ln + j] = (unsigned char)m;
    }
  }
}

__global__ __launch_bounds__(128) void qkv_gemm_kernel(
    const _Float16* __restrict__ xh, const _Float16* __restrict__ wT,
    _Float16* __restrict__ Q, _Float16* __restrict__ K, _Float16* __restrict__ Vt) {
  const int NTG = (3 * Cn) / 64;
  int wave = (blockIdx.x * blockDim.x + threadIdx.x) >> 5;
  int lane = threadIdx.x & 31;
  int mt = wave / NTG, ntg = wave % NTG;
  if (mt >= (Bn * Ln) / 16) return;

  const int lan = lane & 15;
  const int rowA = mt * 16 + lan;
  const int kbA = (lane < 16) ? 0 : 16;
  const int kbB = (lane < 16) ? 0 : 16;
  const _Float16* arow = xh + (size_t)rowA * Cn;
  const _Float16* bcol[4];
#pragma unroll
  for (int f = 0; f < 4; ++f)
    bcol[f] = wT + (size_t)(ntg * 64 + f * 16 + lan) * Cn;

  v8f acc[4] = {};
  for (int kk = 0; kk < Cn; kk += 32) {
    v8h a0 = *(const v8h*)(arow + kk + kbA);
    v8h a1 = *(const v8h*)(arow + kk + kbA + 8);
    v16h av = cat16(a0, a1);
#pragma unroll
    for (int f = 0; f < 4; ++f) {
      v8h b0 = *(const v8h*)(bcol[f] + kk + kbB);
      v8h b1 = *(const v8h*)(bcol[f] + kk + kbB + 8);
      acc[f] = __builtin_amdgcn_wmma_f32_16x16x32_f16(false, av, false, cat16(b0, b1),
                                                      (short)0, acc[f], false, false);
    }
  }

  const int hi = (lane >= 16) ? 8 : 0;
#pragma unroll
  for (int f = 0; f < 4; ++f) {
    const int colB = ntg * 64 + f * 16 + lan;
    const int which = colB >> 9;
    const int c = colB & (Cn - 1);
    const int h = c >> 6, d = c & 63;
#pragma unroll
    for (int r = 0; r < 8; ++r) {
      int m = mt * 16 + r + hi;
      int b = m >> 10, l = m & (Ln - 1);
      size_t bh = (size_t)(b * Hn + h);
      _Float16 val = (_Float16)acc[f][r];
      if (which == 0)      Q[(bh * Ln + l) * Dn + d] = val;
      else if (which == 1) K[(bh * Ln + l) * Dn + d] = val;
      else                 Vt[(bh * Dn + d) * Ln + l] = val;
    }
  }
}

__global__ __launch_bounds__(128) void attn_kernel(
    const _Float16* __restrict__ Q, const _Float16* __restrict__ Kt,
    const _Float16* __restrict__ Vt, const unsigned char* __restrict__ bias_pack,
    const float* __restrict__ gamma, _Float16* __restrict__ aoh) {
  __shared__ float lut[32];
  __shared__ __align__(16) _Float16 pbuf[4][16 * 32];

  const int bh = blockIdx.x >> 4;
  const int tgrp = blockIdx.x & 15;
  const int b = bh >> 3, h = bh & 7;
  const int tid = threadIdx.x;
  if (tid < 32) {
    float s = 0.0f;
#pragma unroll
    for (int k = 0; k < 5; ++k)
      if ((tid >> k) & 1) s += gamma[h * 5 + k];
    lut[tid] = s;
  }
  __syncthreads();

  const int w = tid >> 5, lane = tid & 31;
  const int it = tgrp * 4 + w;
  const int hi = (lane >= 16) ? 8 : 0;
  const int lan = lane & 15;
  const int kbA = (lane < 16) ? 0 : 16;
  const int kbB = (lane < 16) ? 0 : 16;

  const _Float16* Qp = Q + (size_t)bh * Ln * Dn;
  const _Float16* Kp = Kt + (size_t)bh * Ln * Dn;
  const _Float16* Vp = Vt + (size_t)bh * Dn * Ln;
  const unsigned char* bp = bias_pack + (size_t)b * Ln * Ln;

  const int rowA = it * 16 + lan;
  v16h qa[2];
#pragma unroll
  for (int f = 0; f < 2; ++f) {
    v8h lo = *(const v8h*)(Qp + (size_t)rowA * Dn + 32 * f + kbA);
    v8h hh = *(const v8h*)(Qp + (size_t)rowA * Dn + 32 * f + kbA + 8);
    qa[f] = cat16(lo, hh);
  }

  float mrow[8], lrow[8];
  v8f Oacc[4] = {};
#pragma unroll
  for (int r = 0; r < 8; ++r) { mrow[r] = -3.0e38f; lrow[r] = 0.0f; }

  _Float16* pw = pbuf[w];

  for (int j0 = 0; j0 < Ln; j0 += 32) {
    float sv[2][8];
#pragma unroll
    for (int s = 0; s < 2; ++s) {
      const int j = j0 + 16 * s + lan;
      v8f acc = {};
#pragma unroll
      for (int f = 0; f < 2; ++f) {
        v8h b0 = *(const v8h*)(Kp + (size_t)j * Dn + 32 * f + kbB);
        v8h b1 = *(const v8h*)(Kp + (size_t)j * Dn + 32 * f + kbB + 8);
        acc = __builtin_amdgcn_wmma_f32_16x16x32_f16(false, qa[f], false, cat16(b0, b1),
                                                     (short)0, acc, false, false);
      }
#pragma unroll
      for (int r = 0; r < 8; ++r) {
        int i = it * 16 + r + hi;
        unsigned m = bp[(size_t)i * Ln + j];
        sv[s][r] = acc[r] * 0.125f + lut[m & 31u];
      }
    }

    float alpha[8];
#pragma unroll
    for (int r = 0; r < 8; ++r) {
      float mx = fmaxf(sv[0][r], sv[1][r]);
#pragma unroll
      for (int msk = 1; msk < 16; msk <<= 1) mx = fmaxf(mx, __shfl_xor(mx, msk, 32));
      float mnew = fmaxf(mrow[r], mx);
      alpha[r] = __expf(mrow[r] - mnew);
      mrow[r] = mnew;
      float p0 = __expf(sv[0][r] - mnew);
      float p1 = __expf(sv[1][r] - mnew);
      sv[0][r] = p0; sv[1][r] = p1;
      float ps = p0 + p1;
#pragma unroll
      for (int msk = 1; msk < 16; msk <<= 1) ps += __shfl_xor(ps, msk, 32);
      lrow[r] = lrow[r] * alpha[r] + ps;
    }
#pragma unroll
    for (int dt = 0; dt < 4; ++dt)
#pragma unroll
      for (int r = 0; r < 8; ++r) Oacc[dt][r] *= alpha[r];

#pragma unroll
    for (int s = 0; s < 2; ++s)
#pragma unroll
      for (int r = 0; r < 8; ++r)
        pw[(r + hi) * 32 + 16 * s + lan] = (_Float16)sv[s][r];
    asm volatile("" ::: "memory");
    v8h plo = *(const v8h*)(pw + lan * 32 + kbA);
    v8h phi = *(const v8h*)(pw + lan * 32 + kbA + 8);
    v16h pa = cat16(plo, phi);

#pragma unroll
    for (int dt = 0; dt < 4; ++dt) {
      int d = dt * 16 + lan;
      v8h v0 = *(const v8h*)(Vp + (size_t)d * Ln + j0 + kbB);
      v8h v1 = *(const v8h*)(Vp + (size_t)d * Ln + j0 + kbB + 8);
      Oacc[dt] = __builtin_amdgcn_wmma_f32_16x16x32_f16(false, pa, false, cat16(v0, v1),
                                                        (short)0, Oacc[dt], false, false);
    }
  }

#pragma unroll
  for (int dt = 0; dt < 4; ++dt) {
    int c = h * Dn + dt * 16 + lan;
#pragma unroll
    for (int r = 0; r < 8; ++r) {
      int i = it * 16 + r + hi;
      size_t row = (size_t)b * Ln + i;
      aoh[row * Cn + c] = (_Float16)(Oacc[dt][r] / lrow[r]);
    }
  }
}

__global__ __launch_bounds__(128) void proj_gemm_kernel(
    const _Float16* __restrict__ aoh, const _Float16* __restrict__ wT,
    float* __restrict__ out) {
  const int NTG = Cn / 64;
  int wave = (blockIdx.x * blockDim.x + threadIdx.x) >> 5;
  int lane = threadIdx.x & 31;
  int mt = wave / NTG, ntg = wave % NTG;
  if (mt >= (Bn * Ln) / 16) return;

  const int lan = lane & 15;
  const int rowA = mt * 16 + lan;
  const int kbA = (lane < 16) ? 0 : 16;
  const int kbB = (lane < 16) ? 0 : 16;
  const _Float16* arow = aoh + (size_t)rowA * Cn;
  const _Float16* bcol[4];
#pragma unroll
  for (int f = 0; f < 4; ++f)
    bcol[f] = wT + (size_t)(ntg * 64 + f * 16 + lan) * Cn;

  v8f acc[4] = {};
  for (int kk = 0; kk < Cn; kk += 32) {
    v8h a0 = *(const v8h*)(arow + kk + kbA);
    v8h a1 = *(const v8h*)(arow + kk + kbA + 8);
    v16h av = cat16(a0, a1);
#pragma unroll
    for (int f = 0; f < 4; ++f) {
      v8h b0 = *(const v8h*)(bcol[f] + kk + kbB);
      v8h b1 = *(const v8h*)(bcol[f] + kk + kbB + 8);
      acc[f] = __builtin_amdgcn_wmma_f32_16x16x32_f16(false, av, false, cat16(b0, b1),
                                                      (short)0, acc[f], false, false);
    }
  }

  const int hi = (lane >= 16) ? 8 : 0;
#pragma unroll
  for (int f = 0; f < 4; ++f) {
    const int colB = ntg * 64 + f * 16 + lan;
#pragma unroll
    for (int r = 0; r < 8; ++r) {
      size_t row = (size_t)(mt * 16 + r + hi);
      out[row * Cn + colB] = acc[f][r];
    }
  }
}

extern "C" void kernel_launch(void* const* d_in, const int* in_sizes, int n_in,
                              void* d_out, int out_size, void* d_ws, size_t ws_size,
                              hipStream_t stream) {
  (void)in_sizes; (void)n_in; (void)out_size; (void)ws_size;
  const float* x        = (const float*)d_in[0];
  const float* adj      = (const float*)d_in[1];
  const float* distance = (const float*)d_in[2];
  const float* w_qkv    = (const float*)d_in[3];
  const float* w_proj   = (const float*)d_in[4];
  const float* gamma    = (const float*)d_in[5];
  float* out = (float*)d_out;

  char* wsb = (char*)d_ws;
  size_t off = 0;
  auto nxt = [&](size_t bytes) -> char* {
    char* p = wsb + off;
    off += (bytes + 255) & ~(size_t)255;
    return p;
  };
  _Float16* xh         = (_Float16*)nxt((size_t)Bn * Ln * Cn * 2);
  _Float16* wqkvT      = (_Float16*)nxt((size_t)3 * Cn * Cn * 2);
  _Float16* wprojT     = (_Float16*)nxt((size_t)Cn * Cn * 2);
  unsigned char* adj8  = (unsigned char*)nxt((size_t)Bn * Ln * Ln);
  unsigned char* biasP = (unsigned char*)nxt((size_t)Bn * Ln * Ln);
  _Float16* Qb         = (_Float16*)nxt((size_t)Bn * Hn * Ln * Dn * 2);
  _Float16* Kb         = (_Float16*)nxt((size_t)Bn * Hn * Ln * Dn * 2);
  _Float16* Vt         = (_Float16*)nxt((size_t)Bn * Hn * Ln * Dn * 2);
  _Float16* aoh        = (_Float16*)nxt((size_t)Bn * Ln * Cn * 2);

  {
    int n = Bn * Ln * Cn;
    cvt_f32_to_f16_kernel<<<(n + 255) / 256, 256, 0, stream>>>(x, xh, n);
  }
  {
    int n = Cn * 3 * Cn;
    transpose_to_f16_kernel<<<(n + 255) / 256, 256, 0, stream>>>(w_qkv, wqkvT, Cn, 3 * Cn);
  }
  {
    int n = Cn * Cn;
    transpose_to_f16_kernel<<<(n + 255) / 256, 256, 0, stream>>>(w_proj, wprojT, Cn, Cn);
  }
  {
    int n = Bn * Ln * Ln;
    pack_adj_kernel<<<(n + 255) / 256, 256, 0, stream>>>(adj, adj8, n);
  }
  struct_bias_kernel<<<(Bn * 64 * 32) / 4, 128, 0, stream>>>(adj8, distance, biasP);
  qkv_gemm_kernel<<<(512 * 24) / 4, 128, 0, stream>>>(xh, wqkvT, Qb, Kb, Vt);
  attn_kernel<<<Bn * Hn * 16, 128, 0, stream>>>(Qb, Kb, Vt, biasP, gamma, aoh);
  proj_gemm_kernel<<<(512 * 8) / 4, 128, 0, stream>>>(aoh, wprojT, out);
}
